// IntentSlotJointModel_75892072120353
// MI455X (gfx1250) — compile-verified, matrix operand order checked by reading, not run on hardware
//
#include <hip/hip_runtime.h>
#include <hip/hip_bf16.h>

// ---------------------------------------------------------------------------
// IntentSlotJointModel on gfx1250 (wave32, WMMA f32<=f16 16x16x32)
//
//   B=64, S=128, EMB=ENC=256, N_INTENTS=64, N_SLOTS=128, LEN_INTENT=64
//   out = [ intent_batch (64x64) | slots_batch (8192x128) ]  (f32)
//
// One-hot concat folded into slots epilogue: slots[m,n] =
//   (word_enc[m,:] @ slot_dec_W[:256,n]) + slot_dec_b[n]
//   + slot_dec_W[256 + intent[m/128], n]
// ---------------------------------------------------------------------------

typedef __attribute__((ext_vector_type(16))) _Float16 v16h;
typedef __attribute__((ext_vector_type(8)))  _Float16 v8h;
typedef __attribute__((ext_vector_type(8)))  float    v8f;

#define B_   64
#define S_   128
#define ROWS 8192          // B*S
#define EMB_ 256
#define ENC_ 256
#define NSLOT 128
#define NINT 64

// K-index held by (lane, element e) of a 16-bit A/B fragment (CDNA5 ISA):
// lanes 0-15:  e0..7 -> K 0..7,  e8..15 -> K 16..23
// lanes 16-31: e0..7 -> K 8..15, e8..15 -> K 24..31
__device__ __forceinline__ int kmap_elem(int lane, int e) {
    return ((lane & 16) ? 8 : 0) + (e < 8 ? e : e + 8);
}

// ---------------------------------------------------------------------------
// Pack a row-major f32 weight W[K=256, N] into per-tile wave32 B fragments:
// Bpack[(kt*NT + nt)*512 + lane*16 + e] = (f16) W[(kt*32 + kmap(lane,e)) * N
//                                                 + nt*16 + (lane&15)]
// ---------------------------------------------------------------------------
__global__ void pack_B_kernel(const float* __restrict__ W, _Float16* __restrict__ out,
                              int N, int NT, int total) {
    int idx = blockIdx.x * 256 + threadIdx.x;
    if (idx >= total) return;
    int e    = idx & 15;
    int lane = (idx >> 4) & 31;
    int t    = idx >> 9;            // kt*NT + nt
    int nt   = t % NT;
    int kt   = t / NT;
    int k    = kt * 32 + kmap_elem(lane, e);
    int n    = nt * 16 + (lane & 15);
    out[idx] = (_Float16)W[k * N + n];
}

// ---------------------------------------------------------------------------
// Embedding gather -> f16 row-major activations. grid=8192 blocks x 256 thr.
// ---------------------------------------------------------------------------
__global__ void gather_kernel(const int* __restrict__ token_ids,
                              const float* __restrict__ emb_table,
                              _Float16* __restrict__ emb_f16) {
    int row = blockIdx.x;           // b*S + s
    int e   = threadIdx.x;          // 0..255
    int tok = token_ids[row];
    emb_f16[row * EMB_ + e] = (_Float16)emb_table[tok * EMB_ + e];
}

// sent[b,e] = mean_s emb[b,s,e]   (f32 accumulate)
__global__ void sent_kernel(const _Float16* __restrict__ emb_f16,
                            float* __restrict__ sent) {
    int b = blockIdx.x, e = threadIdx.x;
    float acc = 0.0f;
    for (int s = 0; s < S_; ++s)
        acc += (float)emb_f16[(b * S_ + s) * EMB_ + e];
    sent[b * EMB_ + e] = acc * (1.0f / (float)S_);
}

// sent_enc[b,h] = relu(sent[b,:] @ doc_enc_W[:,h] + doc_enc_b[h])
__global__ void sent_enc_kernel(const float* __restrict__ sent,
                                const float* __restrict__ W,
                                const float* __restrict__ bias,
                                float* __restrict__ sent_enc) {
    int b = blockIdx.x, h = threadIdx.x;
    float acc = bias[h];
    for (int e = 0; e < EMB_; ++e)
        acc += sent[b * EMB_ + e] * W[e * ENC_ + h];
    sent_enc[b * ENC_ + h] = acc > 0.0f ? acc : 0.0f;
}

// intent[b,j] = sent_enc[b,:] @ doc_dec_W[:,j] + doc_dec_b[j]
__global__ void intent_kernel(const float* __restrict__ sent_enc,
                              const float* __restrict__ W,
                              const float* __restrict__ bias,
                              float* __restrict__ out) {
    int b = blockIdx.x, j = threadIdx.x;   // block of 64 threads
    float acc = bias[j];
    for (int h = 0; h < ENC_; ++h)
        acc += sent_enc[b * ENC_ + h] * W[h * NINT + j];
    out[b * NINT + j] = acc;
}

// ---------------------------------------------------------------------------
// WMMA GEMM 1: word_enc = relu(emb_f16[8192,256] @ slot_enc_W + b) -> f16
// One 16x16 output tile per wave; 8 k-tiles of 32 -> 8 v_wmma per wave.
// grid = 1024 blocks x 256 threads (8 waves) = 8192 tiles (512 M x 16 N).
// ---------------------------------------------------------------------------
__global__ void word_enc_kernel(const _Float16* __restrict__ A,      // [8192,256]
                                const _Float16* __restrict__ Bpack,  // NT=16
                                const float* __restrict__ bias,      // [256]
                                _Float16* __restrict__ Wout) {       // [8192,256]
    const int lane = threadIdx.x & 31;
    const int wave = threadIdx.x >> 5;
    const int tile = blockIdx.x * 8 + wave;
    const int nt = tile & 15;
    const int mt = tile >> 4;
    const int NT = 16;

    const int row  = mt * 16 + (lane & 15);
    const int koff = (lane & 16) ? 8 : 0;

    v8f acc = {};
    #pragma unroll
    for (int kt = 0; kt < 8; ++kt) {
        const _Float16* ap = A + row * EMB_ + kt * 32 + koff;
        union { v16h v; v8h h[2]; } a;
        a.h[0] = *(const v8h*)(ap);        // K = base..base+7
        a.h[1] = *(const v8h*)(ap + 16);   // K = base+16..base+23
        v16h bf = *(const v16h*)(Bpack + (kt * NT + nt) * 512 + lane * 16);
        acc = __builtin_amdgcn_wmma_f32_16x16x32_f16(
            false, a.v, false, bf, (short)0, acc, false, false);
    }

    const int n     = nt * 16 + (lane & 15);
    const int mbase = mt * 16 + ((lane & 16) ? 8 : 0);
    const float bn  = bias[n];
    #pragma unroll
    for (int r = 0; r < 8; ++r) {
        float v = acc[r] + bn;
        v = v > 0.0f ? v : 0.0f;
        Wout[(mbase + r) * ENC_ + n] = (_Float16)v;
    }
}

// ---------------------------------------------------------------------------
// WMMA GEMM 2: slots = word_enc @ slot_dec_W[:256] + b + onehot row  (f32 out)
// grid = 512 blocks x 256 threads = 4096 tiles (512 M x 8 N).
// ---------------------------------------------------------------------------
__global__ void slots_kernel(const _Float16* __restrict__ A,        // [8192,256]
                             const _Float16* __restrict__ Bpack,    // NT=8
                             const float* __restrict__ slot_dec_W,  // [320,128]
                             const float* __restrict__ slot_dec_b,  // [128]
                             const int* __restrict__ intents,       // [64]
                             float* __restrict__ out) {             // [8192,128]
    const int lane = threadIdx.x & 31;
    const int wave = threadIdx.x >> 5;
    const int tile = blockIdx.x * 8 + wave;
    const int nt = tile & 7;
    const int mt = tile >> 3;
    const int NT = 8;

    const int row  = mt * 16 + (lane & 15);
    const int koff = (lane & 16) ? 8 : 0;

    v8f acc = {};
    #pragma unroll
    for (int kt = 0; kt < 8; ++kt) {
        const _Float16* ap = A + row * ENC_ + kt * 32 + koff;
        union { v16h v; v8h h[2]; } a;
        a.h[0] = *(const v8h*)(ap);
        a.h[1] = *(const v8h*)(ap + 16);
        v16h bf = *(const v16h*)(Bpack + (kt * NT + nt) * 512 + lane * 16);
        acc = __builtin_amdgcn_wmma_f32_16x16x32_f16(
            false, a.v, false, bf, (short)0, acc, false, false);
    }

    const int n     = nt * 16 + (lane & 15);
    const int mbase = mt * 16 + ((lane & 16) ? 8 : 0);
    const int b     = (mt * 16) >> 7;              // 16 | 128 -> constant per tile
    const float bn  = slot_dec_b[n] + slot_dec_W[(ENC_ + intents[b]) * NSLOT + n];
    #pragma unroll
    for (int r = 0; r < 8; ++r)
        out[(mbase + r) * NSLOT + n] = acc[r] + bn;
}

// ---------------------------------------------------------------------------
extern "C" void kernel_launch(void* const* d_in, const int* in_sizes, int n_in,
                              void* d_out, int out_size, void* d_ws, size_t ws_size,
                              hipStream_t stream) {
    const int*   token_ids   = (const int*)d_in[0];
    const int*   all_intents = (const int*)d_in[1];
    const float* emb_table   = (const float*)d_in[2];
    const float* doc_enc_W   = (const float*)d_in[3];
    const float* doc_enc_b   = (const float*)d_in[4];
    const float* slot_enc_W  = (const float*)d_in[5];
    const float* slot_enc_b  = (const float*)d_in[6];
    const float* doc_dec_W   = (const float*)d_in[7];
    const float* doc_dec_b   = (const float*)d_in[8];
    const float* slot_dec_W  = (const float*)d_in[9];
    const float* slot_dec_b  = (const float*)d_in[10];

    char* ws = (char*)d_ws;
    _Float16* emb_f16  = (_Float16*)(ws + 0);            // 8192*256*2 = 4 MB
    _Float16* word_enc = (_Float16*)(ws + 4194304);      // 4 MB
    float*    sent     = (float*)(ws + 8388608);         // 64 KB
    float*    sent_enc = (float*)(ws + 8454144);         // 64 KB
    _Float16* Bpack1   = (_Float16*)(ws + 8519680);      // 65536 f16 = 128 KB
    _Float16* Bpack2   = (_Float16*)(ws + 8650752);      // 32768 f16 = 64 KB

    float* out_intent = (float*)d_out;                   // [64,64]
    float* out_slots  = out_intent + B_ * NINT;          // [8192,128]

    // Weight packing (f32 -> f16 WMMA B fragments)
    pack_B_kernel<<<65536 / 256, 256, 0, stream>>>(slot_enc_W, Bpack1, 256, 16, 65536);
    pack_B_kernel<<<32768 / 256, 256, 0, stream>>>(slot_dec_W, Bpack2, 128, 8, 32768);

    // Embedding gather + doc path
    gather_kernel<<<ROWS, 256, 0, stream>>>(token_ids, emb_table, emb_f16);
    sent_kernel<<<B_, 256, 0, stream>>>(emb_f16, sent);
    sent_enc_kernel<<<B_, 256, 0, stream>>>(sent, doc_enc_W, doc_enc_b, sent_enc);
    intent_kernel<<<B_, NINT, 0, stream>>>(sent_enc, doc_dec_W, doc_dec_b, out_intent);

    // Slot path (WMMA GEMMs)
    word_enc_kernel<<<1024, 256, 0, stream>>>(emb_f16, Bpack1, slot_enc_b, word_enc);
    slots_kernel<<<512, 256, 0, stream>>>(word_enc, Bpack2, slot_dec_W, slot_dec_b,
                                          all_intents, out_slots);
}
